// InterGraphAttention_20194936226354
// MI455X (gfx1250) — matrix operand order repaired and checked by reading, not compiled, not run on hardware
//
#include <hip/hip_runtime.h>
#include <hip/hip_bf16.h>

typedef __attribute__((ext_vector_type(16))) __bf16 v16bf;
typedef __attribute__((ext_vector_type(8)))  __bf16 v8bf;
typedef __attribute__((ext_vector_type(8)))  float  v8f;

#define HEADS 2
#define CH    32
#define DIN   128
#define DOUT  64   // HEADS*CH
#define NEG_SLOPE 0.2f

static __device__ __forceinline__ void atomicMaxF(float* addr, float val) {
  // sign-aware punning: non-negative floats order as ints, negative as reversed uints
  if (val >= 0.f) atomicMax((int*)addr, __float_as_int(val));
  else            atomicMin((unsigned int*)addr, __float_as_uint(val));
}

// ---------------- prep kernels ----------------

__global__ void elu_to_bf16(const float* __restrict__ x, __bf16* __restrict__ y, long n) {
  long i = (long)blockIdx.x * blockDim.x + threadIdx.x;
  if (i >= n) return;
  float v = x[i];
  float e = (v > 0.f) ? v : (__expf(v) - 1.f);
  y[i] = (__bf16)e;
}

__global__ void transpose_w_bf16(const float* __restrict__ W, __bf16* __restrict__ Wt) {
  int i = blockIdx.x * blockDim.x + threadIdx.x;   // over DOUT*DIN
  if (i >= DOUT * DIN) return;
  int n = i / DIN, k = i % DIN;
  Wt[i] = (__bf16)W[k * DOUT + n];                 // Wt[n][k] = W[k][n]
}

__global__ void fill_f32(float* __restrict__ p, long n, float v) {
  long i = (long)blockIdx.x * blockDim.x + threadIdx.x;
  if (i < n) p[i] = v;
}

// ---------------- WMMA GEMM: Y[N,64] = Xbf[N,128] @ W(bf16, pre-transposed [64][128]) ----------------

__global__ void gemm_wmma_bf16(const __bf16* __restrict__ X, const __bf16* __restrict__ Wt,
                               float* __restrict__ Y, int Nrows) {
  const int lane  = threadIdx.x & 31;
  const int wave  = threadIdx.x >> 5;
  const int strip = blockIdx.x * 8 + wave;
  const int m0    = strip * 16;
  if (m0 >= Nrows) return;           // wave-uniform: EXEC stays all-ones for WMMA
  const int half = lane >> 4;        // 0: lanes 0-15, 1: lanes 16-31
  const int lr   = lane & 15;

  v8f acc[4] = {v8f{}, v8f{}, v8f{}, v8f{}};
  const __bf16* xrow = X + (long)(m0 + lr) * DIN;

#pragma unroll
  for (int k0 = 0; k0 < DIN; k0 += 32) {
    // A tile 16x32: lane half0 holds K {k0..k0+7, k0+16..k0+23}; half1 holds {+8..+15, +24..+31}
    v8bf alo = *(const v8bf*)(xrow + k0 + half * 8);
    v8bf ahi = *(const v8bf*)(xrow + k0 + 16 + half * 8);
    v16bf a;
#pragma unroll
    for (int i = 0; i < 8; i++) { a[i] = alo[i]; a[8 + i] = ahi[i]; }

#pragma unroll
    for (int t = 0; t < 4; t++) {
      // B tile 32x16: lane column = lr, half0 holds K {k0..k0+7, k0+16..k0+23}; half1 holds {+8..+15, +24..+31}
      const __bf16* wcol = Wt + (long)(t * 16 + lr) * DIN + k0 + half * 8;
      v8bf blo = *(const v8bf*)(wcol);
      v8bf bhi = *(const v8bf*)(wcol + 16);
      v16bf b;
#pragma unroll
      for (int i = 0; i < 8; i++) { b[i] = blo[i]; b[8 + i] = bhi[i]; }

      acc[t] = __builtin_amdgcn_wmma_f32_16x16x32_bf16(
          false, a, false, b, (short)0, acc[t], false, false);
    }
  }

  // C/D layout: VGPR r -> row m0 + r + half*8, col = t*16 + lr
#pragma unroll
  for (int t = 0; t < 4; t++) {
#pragma unroll
    for (int r = 0; r < 8; r++) {
      int row = m0 + r + half * 8;
      int col = t * 16 + lr;
      Y[(long)row * DOUT + col] = acc[t][r];
    }
  }
}

// ---------------- attention logits: al[n,h] = dot(Y[n, h*32 : h*32+32], att[h,:]) ----------------

__global__ void attn_logits(const float* __restrict__ Y, const float* __restrict__ att,
                            float* __restrict__ al, int Nrows) {
  int i = blockIdx.x * blockDim.x + threadIdx.x;
  if (i >= Nrows * HEADS) return;
  int n = i >> 1, h = i & 1;
  const float* y = Y + (long)n * DOUT + h * CH;
  const float* a = att + h * CH;
  float s = 0.f;
#pragma unroll
  for (int c = 0; c < CH; c++) s += y[c] * a[c];
  al[i] = s;
}

// ---------------- edge passes (segment softmax + aggregate) ----------------

__global__ void edge_alpha_max(const int* __restrict__ srcI, const int* __restrict__ dstI,
                               int E, int E2,
                               const float* __restrict__ als, const float* __restrict__ ald,
                               float* __restrict__ alpha_buf, float* __restrict__ mmax) {
  int e = blockIdx.x * blockDim.x + threadIdx.x;
  if (e >= E2) return;
  int s = (e < E) ? srcI[e] : (e - E);
  int d = (e < E) ? dstI[e] : (e - E);
#pragma unroll
  for (int h = 0; h < HEADS; h++) {
    float a = als[s * HEADS + h] + ald[d * HEADS + h];
    a = (a > 0.f) ? a : NEG_SLOPE * a;
    alpha_buf[(long)e * HEADS + h] = a;
    atomicMaxF(&mmax[d * HEADS + h], a);
  }
}

__global__ void edge_exp_sum(const int* __restrict__ srcI, const int* __restrict__ dstI,
                             int E, int E2,
                             float* __restrict__ alpha_buf,
                             const float* __restrict__ mmax, float* __restrict__ ssum) {
  int e = blockIdx.x * blockDim.x + threadIdx.x;
  if (e >= E2) return;
  int d = (e < E) ? dstI[e] : (e - E);
#pragma unroll
  for (int h = 0; h < HEADS; h++) {
    float ev = __expf(alpha_buf[(long)e * HEADS + h] - mmax[d * HEADS + h]);
    alpha_buf[(long)e * HEADS + h] = ev;
    atomicAdd(&ssum[d * HEADS + h], ev);
  }
}

__global__ void edge_aggregate(const int* __restrict__ srcI, const int* __restrict__ dstI,
                               int E, int E2,
                               const float* __restrict__ alpha_buf, const float* __restrict__ ssum,
                               const float* __restrict__ xs, float* __restrict__ out) {
  long tid = (long)blockIdx.x * blockDim.x + threadIdx.x;
  int e  = (int)(tid >> 6);
  int ch = (int)(tid & 63);
  if (e >= E2) return;
  int s = (e < E) ? srcI[e] : (e - E);
  int d = (e < E) ? dstI[e] : (e - E);
  int h = ch >> 5;
  float w = alpha_buf[(long)e * HEADS + h] / (ssum[d * HEADS + h] + 1e-16f);
  atomicAdd(&out[(long)d * DOUT + ch], w * xs[(long)s * DOUT + ch]);
}

__global__ void add_bias(float* __restrict__ out, const float* __restrict__ bias, long n) {
  long i = (long)blockIdx.x * blockDim.x + threadIdx.x;
  if (i >= n) return;
  out[i] += bias[i & 63];
}

// ---------------- host-side launch ----------------

static inline int gblk(long n, int b) { return (int)((n + b - 1) / b); }

extern "C" void kernel_launch(void* const* d_in, const int* in_sizes, int n_in,
                              void* d_out, int out_size, void* d_ws, size_t ws_size,
                              hipStream_t stream) {
  const float* h_x     = (const float*)d_in[0];
  const float* t_x     = (const float*)d_in[1];
  const int*   ei      = (const int*)d_in[2];
  const float* W_src   = (const float*)d_in[3];
  const float* W_dst   = (const float*)d_in[4];
  const float* att_src = (const float*)d_in[5];
  const float* att_dst = (const float*)d_in[6];
  const float* bias    = (const float*)d_in[7];

  const int Nh = in_sizes[0] / DIN;
  const int Nt = in_sizes[1] / DIN;
  const int E  = in_sizes[2] / 2;
  const int L  = (Nh < Nt) ? Nh : Nt;
  const int E2 = E + L;

  const int* srcA = ei;       // edge_index row 0
  const int* dstA = ei + E;   // edge_index row 1

  // ---- workspace bump allocator ----
  char* ws = (char*)d_ws;
  size_t off = 0;
  auto alloc = [&](size_t bytes) -> void* {
    void* p = ws + off;
    off += (bytes + 255) & ~(size_t)255;
    return p;
  };
  __bf16* h_bf  = (__bf16*)alloc((size_t)Nh * DIN * sizeof(__bf16));
  __bf16* t_bf  = (__bf16*)alloc((size_t)Nt * DIN * sizeof(__bf16));
  __bf16* Ws_t  = (__bf16*)alloc((size_t)DOUT * DIN * sizeof(__bf16));
  __bf16* Wd_t  = (__bf16*)alloc((size_t)DOUT * DIN * sizeof(__bf16));
  float*  xs1   = (float*)alloc((size_t)Nh * DOUT * sizeof(float)); // h_in @ W_src
  float*  xd1   = (float*)alloc((size_t)Nt * DOUT * sizeof(float)); // t_in @ W_dst
  float*  xs2   = (float*)alloc((size_t)Nt * DOUT * sizeof(float)); // t_in @ W_src
  float*  xd2   = (float*)alloc((size_t)Nh * DOUT * sizeof(float)); // h_in @ W_dst
  float*  als1  = (float*)alloc((size_t)Nh * HEADS * sizeof(float));
  float*  ald1  = (float*)alloc((size_t)Nt * HEADS * sizeof(float));
  float*  als2  = (float*)alloc((size_t)Nt * HEADS * sizeof(float));
  float*  ald2  = (float*)alloc((size_t)Nh * HEADS * sizeof(float));
  float*  m1    = (float*)alloc((size_t)Nt * HEADS * sizeof(float));
  float*  s1    = (float*)alloc((size_t)Nt * HEADS * sizeof(float));
  float*  m2    = (float*)alloc((size_t)Nh * HEADS * sizeof(float));
  float*  s2    = (float*)alloc((size_t)Nh * HEADS * sizeof(float));
  float*  al1   = (float*)alloc((size_t)E2 * HEADS * sizeof(float)); // per-edge alpha dir1
  float*  al2   = (float*)alloc((size_t)E2 * HEADS * sizeof(float)); // per-edge alpha dir2
  (void)ws_size;

  float* out_h = (float*)d_out;                 // h_rep first (reference return order)
  float* out_t = out_h + (size_t)Nh * DOUT;     // then t_rep

  const int B = 256;
  const float NEG_INF = -__builtin_inff();

  // 1) ELU + bf16 convert
  elu_to_bf16<<<gblk((long)Nh * DIN, B), B, 0, stream>>>(h_x, h_bf, (long)Nh * DIN);
  elu_to_bf16<<<gblk((long)Nt * DIN, B), B, 0, stream>>>(t_x, t_bf, (long)Nt * DIN);

  // 2) weight transpose + bf16
  transpose_w_bf16<<<gblk(DOUT * DIN, B), B, 0, stream>>>(W_src, Ws_t);
  transpose_w_bf16<<<gblk(DOUT * DIN, B), B, 0, stream>>>(W_dst, Wd_t);

  // 3) four WMMA GEMMs (8 waves/block = 128 rows/block)
  int gb_h = gblk(((long)Nh + 15) / 16, 8);
  int gb_t = gblk(((long)Nt + 15) / 16, 8);
  gemm_wmma_bf16<<<gb_h, 256, 0, stream>>>(h_bf, Ws_t, xs1, Nh);
  gemm_wmma_bf16<<<gb_t, 256, 0, stream>>>(t_bf, Wd_t, xd1, Nt);
  gemm_wmma_bf16<<<gb_t, 256, 0, stream>>>(t_bf, Ws_t, xs2, Nt);
  gemm_wmma_bf16<<<gb_h, 256, 0, stream>>>(h_bf, Wd_t, xd2, Nh);

  // 4) attention logits
  attn_logits<<<gblk((long)Nh * HEADS, B), B, 0, stream>>>(xs1, att_src, als1, Nh);
  attn_logits<<<gblk((long)Nt * HEADS, B), B, 0, stream>>>(xd1, att_dst, ald1, Nt);
  attn_logits<<<gblk((long)Nt * HEADS, B), B, 0, stream>>>(xs2, att_src, als2, Nt);
  attn_logits<<<gblk((long)Nh * HEADS, B), B, 0, stream>>>(xd2, att_dst, ald2, Nh);

  // 5) init output / softmax state
  fill_f32<<<gblk((long)out_size, B), B, 0, stream>>>((float*)d_out, (long)out_size, 0.f);
  fill_f32<<<gblk((long)Nt * HEADS, B), B, 0, stream>>>(m1, (long)Nt * HEADS, NEG_INF);
  fill_f32<<<gblk((long)Nt * HEADS, B), B, 0, stream>>>(s1, (long)Nt * HEADS, 0.f);
  fill_f32<<<gblk((long)Nh * HEADS, B), B, 0, stream>>>(m2, (long)Nh * HEADS, NEG_INF);
  fill_f32<<<gblk((long)Nh * HEADS, B), B, 0, stream>>>(s2, (long)Nh * HEADS, 0.f);

  // 6) direction 1: h -> t (dst = t nodes), produces t_rep
  edge_alpha_max<<<gblk(E2, B), B, 0, stream>>>(srcA, dstA, E, E2, als1, ald1, al1, m1);
  edge_exp_sum <<<gblk(E2, B), B, 0, stream>>>(srcA, dstA, E, E2, al1, m1, s1);
  edge_aggregate<<<gblk((long)E2 * 64, B), B, 0, stream>>>(srcA, dstA, E, E2, al1, s1, xs1, out_t);

  // 7) direction 2: t -> h (edge roles swapped), produces h_rep
  edge_alpha_max<<<gblk(E2, B), B, 0, stream>>>(dstA, srcA, E, E2, als2, ald2, al2, m2);
  edge_exp_sum <<<gblk(E2, B), B, 0, stream>>>(dstA, srcA, E, E2, al2, m2, s2);
  edge_aggregate<<<gblk((long)E2 * 64, B), B, 0, stream>>>(dstA, srcA, E, E2, al2, s2, xs2, out_h);

  // 8) bias
  add_bias<<<gblk((long)out_size, B), B, 0, stream>>>((float*)d_out, bias, (long)out_size);
}
